// ExplicitGraphAttentionLearning_28132035789273
// MI455X (gfx1250) — matrix operand order repaired and checked by reading, not compiled, not run on hardware
//
#include <hip/hip_runtime.h>
#include <cstdint>

typedef _Float16 v8h  __attribute__((ext_vector_type(8)));
typedef _Float16 v16h __attribute__((ext_vector_type(16)));
typedef float    v8f  __attribute__((ext_vector_type(8)));
typedef float    v4f  __attribute__((ext_vector_type(4)));
typedef int      v4i  __attribute__((ext_vector_type(4)));

static __device__ __forceinline__ v16h cat8(v8h lo, v8h hi) {
  return __builtin_shufflevector(lo, hi, 0,1,2,3,4,5,6,7,8,9,10,11,12,13,14,15);
}

#define NB 4
#define NN 2000
#define NPAD 2016
#define DD 128
#define NH 4
#define CC 32
#define NWORDS 64     // 2048 bits >= 2000, padded
#define NCHUNK 63     // ceil(2000/32)
#define MROWS (NB*NN) // 8000

// ---- CDNA5 async global->LDS staging (guarded; falls back to vector copy) --
#if __has_builtin(__builtin_amdgcn_global_load_async_to_lds_b128) && \
    __has_builtin(__builtin_amdgcn_s_wait_asynccnt)
#define USE_ASYNC_LDS 1
#else
#define USE_ASYNC_LDS 0
#endif

static __device__ __forceinline__ void cp_lds_b128(void* lds_dst, const void* gsrc) {
#if USE_ASYNC_LDS
  __builtin_amdgcn_global_load_async_to_lds_b128(
      (__attribute__((address_space(1))) v4i*)(void*)gsrc,
      (__attribute__((address_space(3))) v4i*)lds_dst, 0, 0);
#else
  *(v4f*)lds_dst = *(const v4f*)gsrc;
#endif
}
static __device__ __forceinline__ void async_join() {
#if USE_ASYNC_LDS
  __builtin_amdgcn_s_wait_asynccnt(0);
#endif
  __syncthreads();
}

// ---------------- conversions ----------------
__global__ void k_cvt_x(const float* __restrict__ xa, _Float16* __restrict__ xp) {
  int idx = blockIdx.x * blockDim.x + threadIdx.x;
  if (idx >= MROWS * 160) return;
  int r = idx / 160, k = idx % 160;
  int b = r / NN, n = r % NN;
  float v = 0.f;
  if (k < 158) v = xa[((size_t)(b * 8 + 7) * NN + n) * 158 + k];
  xp[idx] = (_Float16)v;
}

// weights [K,128] -> transposed f16 [128, Kp], zero-padded K->Kp
__global__ void k_cvt_wT(const float* __restrict__ w, _Float16* __restrict__ wT,
                         int K, int Kp) {
  int idx = blockIdx.x * blockDim.x + threadIdx.x;
  if (idx >= 128 * Kp) return;
  int n = idx / Kp, k = idx % Kp;
  wT[idx] = (_Float16)((k < K) ? w[(size_t)k * 128 + n] : 0.f);
}

__global__ void k_zpad(_Float16* __restrict__ xT) {
  int idx = blockIdx.x * blockDim.x + threadIdx.x; // NB*128*(NPAD-NN)
  if (idx >= NB * 128 * (NPAD - NN)) return;
  int bc = idx / (NPAD - NN), p = idx % (NPAD - NN);
  xT[(size_t)bc * NPAD + NN + p] = (_Float16)0.f;
}

// ---------------- mask bit-pack (reads 64MB sector_graph once) -------------
__global__ void k_pack(const float* __restrict__ sg, uint32_t* __restrict__ mb) {
  const int row = blockIdx.x;               // b*NN + i
  const int b = row / NN, i = row % NN;
  const int lane = threadIdx.x & 31, wv = threadIdx.x >> 5;
  const float* srow = sg + (size_t)b * NN * NN + (size_t)i * NN;
  for (int w = wv; w < NWORDS; w += 8) {
    int j = w * 32 + lane;
    bool pred = (j < NN) && ((srow[j] > 0.f) || (j == i));
    uint32_t word = __builtin_amdgcn_ballot_w32(pred);
    if (lane == 0) mb[(size_t)row * NWORDS + w] = word;
  }
}

// ---------------- generic WMMA GEMM: [M,Kp]f16 x [Kp,128]f16(T) -> f32 ------
// BT is stored transposed: [128][Kp]. A tile staged via async->LDS (shared by
// all 8 waves). Optional bias add and optional transposed-f16 output.
__global__ __launch_bounds__(256)
void k_gemm(const _Float16* __restrict__ A, const _Float16* __restrict__ BT,
            int Kp, const float* __restrict__ bias,
            float* __restrict__ C, _Float16* __restrict__ outT) {
  __shared__ __align__(16) _Float16 shA[16 * 160];
  const int t = threadIdx.x;
  const int lane = t & 31;
  const int wv = t >> 5;
  const int m0 = blockIdx.x * 16;
  // stage A tile (16 rows x Kp halfs, rows contiguous) into LDS in b128 units
  for (int q = t; q < 2 * Kp; q += 256)
    cp_lds_b128(shA + q * 8, A + (size_t)m0 * Kp + q * 8);
  async_join();

  const int n0 = wv * 16;
  const int col = n0 + (lane & 15);
  const int loffA = (lane >= 16) ? 8 : 0;
  const int loffB = (lane >= 16) ? 16 : 0;
  const int arow = (lane & 15) * Kp;
  v8f acc = {};
  for (int k0 = 0; k0 < Kp; k0 += 32) {
    const v8h* pa = (const v8h*)(shA + arow + k0 + loffB);
    v16h a = cat8(pa[0], pa[1]);               // K = e + loffB
    v16h b = *(const v16h*)(BT + (size_t)col * Kp + k0 + loffB); // K = e + loffB
    acc = __builtin_amdgcn_wmma_f32_16x16x32_f16(false, a, false, b,
                                                 (short)0, acc, false, false);
  }
  const int rbase = m0 + ((lane >= 16) ? 8 : 0);
#pragma unroll
  for (int r = 0; r < 8; ++r) {
    int row = rbase + r;
    float v = acc[r];
    if (bias) v += bias[col];
    C[(size_t)row * 128 + col] = v;
    if (outT) {
      int b_ = row / NN, n_ = row % NN;       // tiles never straddle b (NN%16==0)
      outT[((size_t)(b_ * 128 + col)) * NPAD + n_] = (_Float16)v;
    }
  }
}

// ---------------- bias + LayerNorm + ReLU (row of 128) ----------------------
__global__ void k_ln(const float* __restrict__ tmp, const float* __restrict__ b_in,
                     const float* __restrict__ g, const float* __restrict__ bta,
                     float* __restrict__ h32, _Float16* __restrict__ h16) {
  __shared__ float sh[4];
  const int row = blockIdx.x, t = threadIdx.x, lane = t & 31, wv = t >> 5;
  float v = tmp[(size_t)row * 128 + t] + b_in[t];
  float s = v;
  for (int off = 16; off; off >>= 1) s += __shfl_xor(s, off, 32);
  if (lane == 0) sh[wv] = s;
  __syncthreads();
  float mean = (sh[0] + sh[1] + sh[2] + sh[3]) * (1.0f / 128.0f);
  __syncthreads();
  float d = v - mean;
  float q = d * d;
  for (int off = 16; off; off >>= 1) q += __shfl_xor(q, off, 32);
  if (lane == 0) sh[wv] = q;
  __syncthreads();
  float var = (sh[0] + sh[1] + sh[2] + sh[3]) * (1.0f / 128.0f);
  float o = fmaxf(d * rsqrtf(var + 1e-5f) * g[t] + bta[t], 0.f);
  h32[(size_t)row * 128 + t] = o;
  h16[(size_t)row * 128 + t] = (_Float16)o;
}

// ---------------- attention logit vectors e_src/e_dst [B,H,N] ---------------
__global__ void k_edot(const float* __restrict__ x, const float* __restrict__ as_,
                       const float* __restrict__ ad_, float* __restrict__ e_src,
                       float* __restrict__ e_dst) {
  const int row = blockIdx.x;                 // b*NN + n
  const int t = threadIdx.x;                  // 0..127 ; head = t>>5
  const int lane = t & 31, h = t >> 5;
  float xv = x[(size_t)row * 128 + t];
  float vs = xv * as_[t], vd = xv * ad_[t];
  for (int off = 16; off; off >>= 1) {
    vs += __shfl_xor(vs, off, 32);
    vd += __shfl_xor(vd, off, 32);
  }
  if (lane == 0) {
    int b = row / NN, n = row % NN;
    e_src[(size_t)(b * NH + h) * NN + n] = vs;
    e_dst[(size_t)(b * NH + h) * NN + n] = vd;
  }
}

// ---------------- masked-softmax row stats (m, 1/sum) -----------------------
// leaky_relu is monotone => row max = lrelu(e_dst + max_masked e_src)
__global__ void k_stats(const uint32_t* __restrict__ mb,
                        const float* __restrict__ e_src, const float* __restrict__ e_dst,
                        float* __restrict__ m_arr, float* __restrict__ invl) {
  const int row = blockIdx.x;                 // b*NN + i
  const int b = row / NN, i = row % NN;
  const int lane = threadIdx.x & 31, h = threadIdx.x >> 5;
  const size_t eb = (size_t)(b * NH + h) * NN;
  const uint32_t* mrow = mb + (size_t)row * NWORDS;
  float mx = -3.0e38f;
  for (int w = lane; w < NCHUNK; w += 32) {
    uint32_t word = mrow[w];
    while (word) {
      int bit = __builtin_ctz(word);
      word &= word - 1;
      mx = fmaxf(mx, e_src[eb + w * 32 + bit]);
    }
  }
  for (int off = 16; off; off >>= 1) mx = fmaxf(mx, __shfl_xor(mx, off, 32));
  const float ed = e_dst[eb + i];
  float m = ed + mx;
  m = (m > 0.f) ? m : 0.2f * m;
  float sum = 0.f;
  for (int w = lane; w < NCHUNK; w += 32) {
    uint32_t word = mrow[w];
    while (word) {
      int bit = __builtin_ctz(word);
      word &= word - 1;
      float s = ed + e_src[eb + w * 32 + bit];
      s = (s > 0.f) ? s : 0.2f * s;
      sum += __expf(s - m);
    }
  }
  for (int off = 16; off; off >>= 1) sum += __shfl_xor(sum, off, 32);
  if (lane == 0) { m_arr[eb + i] = m; invl[eb + i] = 1.0f / sum; }
}

// ---------------- WMMA attention aggregation: out = alpha @ x ---------------
// block = 8 waves: wave -> (head h = wv&3, col half c0 = (wv>>2)*16)
// grid  = NB * (NN/16) row-tiles. alpha built on the fly in A layout.
// e_src table for this batch (4 heads x 2000 = 32KB) staged async into LDS.
__global__ __launch_bounds__(256)
void k_agg(const uint32_t* __restrict__ mb,
           const float* __restrict__ e_src, const float* __restrict__ e_dst,
           const float* __restrict__ m_arr, const float* __restrict__ invl,
           const _Float16* __restrict__ xT,   // [NB][128][NPAD] f16, rows>=NN zero
           float* __restrict__ agg) {         // [MROWS][128]
  __shared__ __align__(16) float shE[NH * NN]; // 32KB
  const int lane = threadIdx.x & 31;
  const int wv = threadIdx.x >> 5;
  const int h = wv & 3;
  const int c0 = (wv >> 2) * 16;
  const int b = blockIdx.x / (NN / 16);
  const int i0 = (blockIdx.x % (NN / 16)) * 16;

  // stage e_src[b, :, :] into LDS in b128 units (8000 floats = 2000 units)
  {
    const float* eg = e_src + (size_t)b * NH * NN;
    for (int q = threadIdx.x; q < (NH * NN) / 4; q += 256)
      cp_lds_b128(shE + q * 4, eg + q * 4);
  }
  async_join();

  const int iA = i0 + (lane & 15);
  const size_t eb = (size_t)(b * NH + h) * NN;
  const float edst_i = e_dst[eb + iA];
  const float m_i = m_arr[eb + iA];
  const float il_i = invl[eb + iA];
  const int col = h * CC + c0 + (lane & 15);
  const _Float16* xcol = xT + ((size_t)(b * 128 + col)) * NPAD;
  const uint32_t* mrow = mb + ((size_t)(b * NN + iA)) * NWORDS;
  const float* eh = shE + h * NN;
  const int loffA = (lane >= 16) ? 8 : 0;
  const int loffB = (lane >= 16) ? 16 : 0;
  v8f acc = {};
  for (int jc = 0; jc < NCHUNK; ++jc) {
    const int j0 = jc * 32;
    if (jc + 1 < NCHUNK) __builtin_prefetch(xcol + j0 + 32, 0, 1);
    const uint32_t word = mrow[jc];
    v16h a;
#pragma unroll
    for (int e = 0; e < 16; ++e) {
      int K = e + loffB;  // contiguous K order, matches B
      float alpha = 0.f;
      if ((word >> K) & 1u) {
        float s = edst_i + eh[j0 + K];
        s = (s > 0.f) ? s : 0.2f * s;
        alpha = __expf(s - m_i) * il_i;
      }
      a[e] = (_Float16)alpha;
    }
    v16h bm = *(const v16h*)(xcol + j0 + loffB); // K-striped B layout
    acc = __builtin_amdgcn_wmma_f32_16x16x32_f16(false, a, false, bm,
                                                 (short)0, acc, false, false);
  }
  const int rbase = i0 + ((lane >= 16) ? 8 : 0);
#pragma unroll
  for (int r = 0; r < 8; ++r)
    agg[((size_t)(b * NN + rbase + r)) * 128 + col] = acc[r];
}

// ---------------- layer-0 epilogue: h1 = h + elu(agg + bias0) ---------------
__global__ void k_epi0(const float* __restrict__ h, const float* __restrict__ agg,
                       const float* __restrict__ bias0, float* __restrict__ h1,
                       _Float16* __restrict__ h1h) {
  int idx = blockIdx.x * blockDim.x + threadIdx.x;   // < MROWS*128
  int t = idx & 127;
  float v = agg[idx] + bias0[t];
  float e = (v > 0.f) ? v : (__expf(v) - 1.0f);
  float o = h[idx] + e;
  h1[idx] = o;
  h1h[idx] = (_Float16)o;
}

// ---------------- layer-1 epilogue: head mean + bias1 -> f16 [MROWS,32] -----
__global__ void k_mean(const float* __restrict__ agg, const float* __restrict__ bias1,
                       _Float16* __restrict__ out16) {
  int idx = blockIdx.x * blockDim.x + threadIdx.x;   // < MROWS*32
  int r = idx >> 5, c = idx & 31;
  const float* p = agg + (size_t)r * 128 + c;
  float v = 0.25f * (p[0] + p[32] + p[64] + p[96]) + bias1[c];
  out16[idx] = (_Float16)v;
}

extern "C" void kernel_launch(void* const* d_in, const int* in_sizes, int n_in,
                              void* d_out, int out_size, void* d_ws, size_t ws_size,
                              hipStream_t stream) {
  const float* x_alpha = (const float*)d_in[0];
  const float* sector  = (const float*)d_in[1];
  const float* W_in    = (const float*)d_in[2];
  const float* b_in    = (const float*)d_in[3];
  const float* ln_g    = (const float*)d_in[4];
  const float* ln_b    = (const float*)d_in[5];
  const float* W0      = (const float*)d_in[6];
  const float* as0     = (const float*)d_in[7];
  const float* ad0     = (const float*)d_in[8];
  const float* bias0   = (const float*)d_in[9];
  const float* W1      = (const float*)d_in[10];
  const float* as1     = (const float*)d_in[11];
  const float* ad1     = (const float*)d_in[12];
  const float* bias1   = (const float*)d_in[13];
  const float* W_out   = (const float*)d_in[14];
  const float* b_out   = (const float*)d_in[15];
  float* out = (float*)d_out;

  char* ws = (char*)d_ws;
  size_t off = 0;
  auto alloc = [&](size_t bytes) -> void* {
    void* p = ws + off;
    off = (off + bytes + 255) & ~(size_t)255;
    return p;
  };
  _Float16* xp     = (_Float16*)alloc((size_t)MROWS * 160 * 2);
  _Float16* WinT   = (_Float16*)alloc((size_t)128 * 160 * 2);
  _Float16* W0T    = (_Float16*)alloc((size_t)128 * 128 * 2);
  _Float16* W1T    = (_Float16*)alloc((size_t)128 * 128 * 2);
  _Float16* WoT    = (_Float16*)alloc((size_t)128 * 32 * 2);
  float*    tmp1   = (float*)alloc((size_t)MROWS * 128 * 4);
  float*    hbuf   = (float*)alloc((size_t)MROWS * 128 * 4);
  _Float16* h16    = (_Float16*)alloc((size_t)MROWS * 128 * 2);
  float*    xbuf   = (float*)alloc((size_t)MROWS * 128 * 4);   // x0 / x1 (reused)
  _Float16* xT     = (_Float16*)alloc((size_t)NB * 128 * NPAD * 2);
  float*    e_src  = (float*)alloc((size_t)NB * NH * NN * 4);
  float*    e_dst  = (float*)alloc((size_t)NB * NH * NN * 4);
  float*    m_arr  = (float*)alloc((size_t)NB * NH * NN * 4);
  float*    invl   = (float*)alloc((size_t)NB * NH * NN * 4);
  uint32_t* mb     = (uint32_t*)alloc((size_t)MROWS * NWORDS * 4);
  float*    aggb   = (float*)alloc((size_t)MROWS * 128 * 4);   // reused both layers
  float*    h1buf  = (float*)alloc((size_t)MROWS * 128 * 4);
  _Float16* h1h    = (_Float16*)alloc((size_t)MROWS * 128 * 2);
  _Float16* o116   = (_Float16*)alloc((size_t)MROWS * 32 * 2);

  // conversions + mask packing
  k_cvt_x<<<(MROWS * 160 + 255) / 256, 256, 0, stream>>>(x_alpha, xp);
  k_cvt_wT<<<(128 * 160 + 255) / 256, 256, 0, stream>>>(W_in, WinT, 158, 160);
  k_cvt_wT<<<(128 * 128 + 255) / 256, 256, 0, stream>>>(W0, W0T, 128, 128);
  k_cvt_wT<<<(128 * 128 + 255) / 256, 256, 0, stream>>>(W1, W1T, 128, 128);
  k_cvt_wT<<<(128 * 32 + 255) / 256, 256, 0, stream>>>(W_out, WoT, 32, 32);
  k_zpad<<<(NB * 128 * (NPAD - NN) + 255) / 256, 256, 0, stream>>>(xT);
  k_pack<<<MROWS, 256, 0, stream>>>(sector, mb);

  // input GEMM + LN + ReLU
  k_gemm<<<MROWS / 16, 256, 0, stream>>>(xp, WinT, 160, nullptr, tmp1, nullptr);
  k_ln<<<MROWS, 128, 0, stream>>>(tmp1, b_in, ln_g, ln_b, hbuf, h16);

  // ---- GAT layer 0 (concat) ----
  k_gemm<<<MROWS / 16, 256, 0, stream>>>(h16, W0T, 128, nullptr, xbuf, xT);
  k_edot<<<MROWS, 128, 0, stream>>>(xbuf, as0, ad0, e_src, e_dst);
  k_stats<<<MROWS, 128, 0, stream>>>(mb, e_src, e_dst, m_arr, invl);
  k_agg<<<NB * (NN / 16), 256, 0, stream>>>(mb, e_src, e_dst, m_arr, invl, xT, aggb);
  k_epi0<<<(MROWS * 128) / 256, 256, 0, stream>>>(hbuf, aggb, bias0, h1buf, h1h);

  // ---- GAT layer 1 (head mean) ----
  k_gemm<<<MROWS / 16, 256, 0, stream>>>(h1h, W1T, 128, nullptr, xbuf, xT);
  k_edot<<<MROWS, 128, 0, stream>>>(xbuf, as1, ad1, e_src, e_dst);
  k_stats<<<MROWS, 128, 0, stream>>>(mb, e_src, e_dst, m_arr, invl);
  k_agg<<<NB * (NN / 16), 256, 0, stream>>>(mb, e_src, e_dst, m_arr, invl, xT, aggb);
  k_mean<<<(MROWS * 32) / 256, 256, 0, stream>>>(aggb, bias1, o116);

  // final projection C=32 -> 128 (+ b_out)
  k_gemm<<<MROWS / 16, 256, 0, stream>>>(o116, WoT, 32, b_out, out, nullptr);
}
